// SelfAttention_3135326126545
// MI455X (gfx1250) — matrix operand order repaired and checked by reading, not compiled, not run on hardware
//
#include <hip/hip_runtime.h>

// ---------------------------------------------------------------------------
// CDNA5 (gfx1250) bf16 WMMA self-attention pipeline.
// Shapes: B=16, S=2048, D_IN=256, D_EMB=D_KQ=D_V=512.
// GEMMs + flash attention with async-to-LDS double-buffered K tiles.
// ---------------------------------------------------------------------------

typedef unsigned short u16;
typedef unsigned int   u32;
typedef unsigned long long u64;
typedef __attribute__((ext_vector_type(16))) __bf16 v16bf;
typedef __attribute__((ext_vector_type(8)))  float  v8f;
typedef __attribute__((ext_vector_type(4)))  u32    u32x4;

union FragBF { u32x4 q[2]; v16bf v; };

__device__ __forceinline__ u16 f32_to_bf16(float f) {
  u32 u = __float_as_uint(f);
  u = u + 0x7FFFu + ((u >> 16) & 1u);   // round-to-nearest-even
  return (u16)(u >> 16);
}

__device__ __forceinline__ u32 pack_bf2(float a, float b) {
  return (u32)f32_to_bf16(a) | ((u32)f32_to_bf16(b) << 16);
}

// B-fragment (32x16 K-major) or generic 16 contiguous bf16 elements.
__device__ __forceinline__ v16bf frag_contig(const u16* p) {
  FragBF f;
  f.q[0] = *(const u32x4*)(p);
  f.q[1] = *(const u32x4*)(p + 8);
  return f.v;
}
// A-fragment (16x32): per-lane two 8-element chunks of one row.
__device__ __forceinline__ v16bf frag_2x8(const u16* p0, const u16* p1) {
  FragBF f;
  f.q[0] = *(const u32x4*)(p0);
  f.q[1] = *(const u32x4*)(p1);
  return f.v;
}
// A-fragment built from f32 source (convert on the fly).
__device__ __forceinline__ v16bf frag_f32_2x8(const float* p0, const float* p1) {
  FragBF f;
  f.q[0][0] = pack_bf2(p0[0], p0[1]); f.q[0][1] = pack_bf2(p0[2], p0[3]);
  f.q[0][2] = pack_bf2(p0[4], p0[5]); f.q[0][3] = pack_bf2(p0[6], p0[7]);
  f.q[1][0] = pack_bf2(p1[0], p1[1]); f.q[1][1] = pack_bf2(p1[2], p1[3]);
  f.q[1][2] = pack_bf2(p1[4], p1[5]); f.q[1][3] = pack_bf2(p1[6], p1[7]);
  return f.v;
}

__device__ __forceinline__ v8f wmma_bf16(v16bf a, v16bf b, v8f c) {
  // D(f32 16x16) = A(16x32 bf16) * B(32x16 bf16) + C
  return __builtin_amdgcn_wmma_f32_16x16x32_bf16(
      /*neg_a=*/false, a, /*neg_b=*/false, b,
      /*c_mod=*/(short)0, c, /*reuse_a=*/false, /*reuse_b=*/false);
}

// --- CDNA5 async global->LDS copy (ASYNCcnt-tracked, b128 per lane) --------
// Generic LDS pointers carry the LDS byte offset in their low 32 bits
// (flat aperture mapping), so truncation yields the VDST offset.
__device__ __forceinline__ void async_ld_b128(void* lds_dst, const void* gsrc) {
  u32 lds_off = (u32)(size_t)lds_dst;
  u64 gaddr   = (u64)(size_t)gsrc;
  asm volatile("global_load_async_to_lds_b128 %0, %1, off"
               :: "v"(lds_off), "v"(gaddr)
               : "memory");
}
__device__ __forceinline__ void wait_async_le8() {
  asm volatile("s_wait_asynccnt 0x8" ::: "memory");
}
__device__ __forceinline__ void wait_async_0() {
  asm volatile("s_wait_asynccnt 0x0" ::: "memory");
}

// ---------------------------------------------------------------------------
// Kernel 0: WT[n*K+k] = bf16(W[k*N+n])   (weights transposed so B-frags are
// contiguous row reads)
// ---------------------------------------------------------------------------
__global__ void wt_transpose_kernel(const float* __restrict__ W,
                                    u16* __restrict__ WT, int K, int N) {
  int idx = blockIdx.x * blockDim.x + threadIdx.x;
  if (idx >= K * N) return;
  int n = idx / K, k = idx % K;
  WT[idx] = f32_to_bf16(W[(size_t)k * N + n]);
}

// ---------------------------------------------------------------------------
// Generic WMMA GEMM:  C[M,N] = A[M,K] * BT[N,K]^T
//   AF32:   A is f32 (converted to bf16 on load) else bf16
//   TRANSC: store C transposed per 2048-row batch: C[b][n][s] (for V^T)
// Block = 256 threads = 8 waves (4 M x 2 N), block tile 64x64, wave 16x32.
// ---------------------------------------------------------------------------
template <bool AF32, bool TRANSC>
__global__ __launch_bounds__(256) void gemm_bf16_kernel(
    const void* __restrict__ Av, const u16* __restrict__ BT,
    u16* __restrict__ C, int M, int N, int K) {
  const int wave = threadIdx.x >> 5;
  const int lane = threadIdx.x & 31;
  const int half = lane >> 4;
  const int l16  = lane & 15;
  const int mw = wave & 3, nw = wave >> 2;

  const int rowBase = blockIdx.x * 64 + mw * 16;
  const int colBase = blockIdx.y * 64 + nw * 32;
  const int row = rowBase + l16;

  v8f acc0 = {}, acc1 = {};
  const u16* b0base = BT + (size_t)(colBase + l16) * K;
  const u16* b1base = BT + (size_t)(colBase + 16 + l16) * K;

  for (int kc = 0; kc < K; kc += 32) {
    v16bf a;
    if (AF32) {
      const float* A = (const float*)Av;
      const float* p0 = A + (size_t)row * K + kc + half * 16;
      a = frag_f32_2x8(p0, p0 + 8);
    } else {
      const u16* A = (const u16*)Av;
      const u16* p0 = A + (size_t)row * K + kc + half * 16;
      a = frag_2x8(p0, p0 + 8);
    }
    v16bf b0 = frag_contig(b0base + kc + half * 16);
    v16bf b1 = frag_contig(b1base + kc + half * 16);
    acc0 = wmma_bf16(a, b0, acc0);
    acc1 = wmma_bf16(a, b1, acc1);
  }

#pragma unroll
  for (int r = 0; r < 8; ++r) {
    const int m = rowBase + r + 8 * half;
    const int n0 = colBase + l16;
    if (!TRANSC) {
      C[(size_t)m * N + n0]      = f32_to_bf16(acc0[r]);
      C[(size_t)m * N + n0 + 16] = f32_to_bf16(acc1[r]);
    } else {
      const int b = m >> 11;         // S = 2048
      const int s = m & 2047;
      C[((size_t)b * N + n0) * 2048 + s]        = f32_to_bf16(acc0[r]);
      C[((size_t)b * N + n0 + 16) * 2048 + s]   = f32_to_bf16(acc1[r]);
    }
  }
}

// ---------------------------------------------------------------------------
// Flash attention: per block: one batch b, 32 query rows.
// 8 waves = 2 (M halves of 16 rows) x 4 (contraction split for scores /
// 128-wide D_V slice for PV). Online softmax, unscaled scores (per reference).
// K tiles (32x512 bf16 = 32 KB, contiguous in global) are double-buffered in
// LDS via global_load_async_to_lds_b128 (ASYNCcnt), one tile ahead.
// q,k: bf16 [B,S,512] row-major;  vT: bf16 [B,512,S];  out: f32 [B,S,512].
// ---------------------------------------------------------------------------
__global__ __launch_bounds__(256) void attn_flash_kernel(
    const u16* __restrict__ qg, const u16* __restrict__ kg,
    const u16* __restrict__ vtg, float* __restrict__ out) {
  constexpr int S = 2048, D = 512, NKT = S / 32;
  const int b  = blockIdx.y;
  const int qt = blockIdx.x;                 // query tile of 32 rows
  const int wave = threadIdx.x >> 5;
  const int lane = threadIdx.x & 31;
  const int half = lane >> 4;
  const int l16  = lane & 15;
  const int mw = wave & 1;                   // which 16 query rows
  const int dw = wave >> 1;                  // contraction / D_V slice (0..3)

  __shared__ __align__(16) u16  qs[32 * D];            // 32 KB staged q tile
  __shared__ __align__(16) u16  ks[2][32 * D];         // 64 KB dbl-buffered k
  __shared__ float spart[2][4][16][32];                // partial scores, 16 KB
  __shared__ __align__(16) u16  pbuf[2][16 * 32];      // P tile bf16, 2 KB
  __shared__ float scal[2][16], mrun[2][16], lrun[2][16];

  const size_t kbatch = (size_t)b * S * D;   // k  [S,D]
  const size_t vbatch = (size_t)b * D * S;   // vT [D,S]

  // ---- stage q tile (coalesced b128 copies) ----
  {
    const size_t gbase = ((size_t)b * S + (size_t)qt * 32) * D;
#pragma unroll
    for (int it = 0; it < 8; ++it) {
      int c  = threadIdx.x + it * 256;       // 16-byte chunk id, 2048 total
      int r  = c >> 6;                       // 64 chunks per row
      int cc = c & 63;
      *(u32x4*)&qs[r * D + cc * 8] =
          *(const u32x4*)&qg[gbase + (size_t)r * D + cc * 8];
    }
  }
  // ---- async-stage k tile 0 (tile is 32 KB contiguous in global) ----
  {
    const u16* gsrc = kg + kbatch;           // tile 0
#pragma unroll
    for (int it = 0; it < 8; ++it) {
      int c = threadIdx.x + it * 256;        // b128 chunk id, 2048 total
      async_ld_b128(&ks[0][c * 8], gsrc + (size_t)c * 8);
    }
  }
  if (wave < 2 && lane < 16) { mrun[wave][lane] = -1e30f; lrun[wave][lane] = 0.f; }

  v8f acc[8];
#pragma unroll
  for (int i = 0; i < 8; ++i) acc[i] = v8f{};

  for (int kt = 0; kt < NKT; ++kt) {
    const int cur = kt & 1;

    // -------- stage next k tile, then make ks[cur] visible ---------------
    if (kt + 1 < NKT) {
      const u16* gsrc = kg + kbatch + (size_t)(kt + 1) * 32 * D;
#pragma unroll
      for (int it = 0; it < 8; ++it) {
        int c = threadIdx.x + it * 256;
        async_ld_b128(&ks[cur ^ 1][c * 8], gsrc + (size_t)c * 8);
      }
      wait_async_le8();                      // tile kt done (in-order ASYNCcnt)
    } else {
      wait_async_0();
    }
    __syncthreads();                         // publish ks[cur]; close phase 3

    // -------- phase 1: partial scores over this wave's 128-wide K slice ----
    v8f s0 = {}, s1 = {};
    const int kcB = dw * 128;
#pragma unroll
    for (int c = 0; c < 4; ++c) {
      const int kc = kcB + c * 32;
      const u16* qrow = &qs[(mw * 16 + l16) * D + kc + half * 16];
      v16bf a  = frag_2x8(qrow, qrow + 8);
      v16bf b0 = frag_contig(&ks[cur][l16 * D        + kc + half * 16]);
      v16bf b1 = frag_contig(&ks[cur][(16 + l16) * D + kc + half * 16]);
      s0 = wmma_bf16(a, b0, s0);
      s1 = wmma_bf16(a, b1, s1);
    }
#pragma unroll
    for (int r = 0; r < 8; ++r) {
      spart[mw][dw][r + 8 * half][l16]      = s0[r];
      spart[mw][dw][r + 8 * half][16 + l16] = s1[r];
    }
    if (kt + 1 < NKT)   // pull next tile's v rows toward L2 (global_prefetch_b8)
      __builtin_prefetch(vtg + vbatch + (size_t)(dw * 128 + lane) * S + (kt + 1) * 32, 0, 2);
    __syncthreads();

    // -------- phase 2: online softmax (one wave per M half) ---------------
    if (dw == 0) {
      for (int r2 = 0; r2 < 16; ++r2) {
        float s = spart[mw][0][r2][lane] + spart[mw][1][r2][lane] +
                  spart[mw][2][r2][lane] + spart[mw][3][r2][lane];
        float rmax = s;
#pragma unroll
        for (int o = 16; o > 0; o >>= 1) rmax = fmaxf(rmax, __shfl_xor(rmax, o, 32));
        const float mo = mrun[mw][r2];
        const float mn = fmaxf(mo, rmax);
        const float p  = __expf(s - mn);
        float rsum = p;
#pragma unroll
        for (int o = 16; o > 0; o >>= 1) rsum += __shfl_xor(rsum, o, 32);
        pbuf[mw][r2 * 32 + lane] = f32_to_bf16(p);
        if (lane == 0) {
          const float cfac = __expf(mo - mn);
          scal[mw][r2] = cfac;
          mrun[mw][r2] = mn;
          lrun[mw][r2] = lrun[mw][r2] * cfac + rsum;
        }
      }
    }
    __syncthreads();

    // -------- phase 3: rescale y-acc and accumulate P @ V -----------------
    float cs[8];
#pragma unroll
    for (int r = 0; r < 8; ++r) cs[r] = scal[mw][r + 8 * half];
    const u16* prow = &pbuf[mw][l16 * 32 + half * 16];
    v16bf pa = frag_2x8(prow, prow + 8);
#pragma unroll
    for (int nt = 0; nt < 8; ++nt) {
      const int n = dw * 128 + nt * 16 + l16;
      v16bf bv = frag_contig(vtg + vbatch + (size_t)n * S + kt * 32 + half * 16);
#pragma unroll
      for (int r = 0; r < 8; ++r) acc[nt][r] *= cs[r];
      acc[nt] = wmma_bf16(pa, bv, acc[nt]);
    }
    // (barrier for this phase is at the top of the next iteration)
  }

  // ---- epilogue: divide by softmax denominator, store f32 ----
  float li[8];
#pragma unroll
  for (int r = 0; r < 8; ++r) li[r] = 1.0f / lrun[mw][r + 8 * half];
#pragma unroll
  for (int nt = 0; nt < 8; ++nt) {
#pragma unroll
    for (int r = 0; r < 8; ++r) {
      const int m = qt * 32 + mw * 16 + r + 8 * half;
      const int n = dw * 128 + nt * 16 + l16;
      out[((size_t)b * S + m) * D + n] = acc[nt][r] * li[r];
    }
  }
}

// ---------------------------------------------------------------------------
// Host launcher
// ---------------------------------------------------------------------------
extern "C" void kernel_launch(void* const* d_in, const int* in_sizes, int n_in,
                              void* d_out, int out_size, void* d_ws, size_t ws_size,
                              hipStream_t stream) {
  (void)in_sizes; (void)n_in; (void)out_size; (void)ws_size;
  constexpr int B = 16, S = 2048, D_IN = 256, D = 512;
  constexpr int M = B * S;                   // 32768 rows

  const float* x  = (const float*)d_in[0];
  const float* We = (const float*)d_in[1];
  const float* Wk = (const float*)d_in[2];
  const float* Wq = (const float*)d_in[3];
  const float* Wv = (const float*)d_in[4];
  float* out = (float*)d_out;

  // workspace layout (bf16 as u16), all 16B-aligned
  u16* ws  = (u16*)d_ws;
  u16* WTe = ws;                                   // [512][256]
  u16* WTk = WTe + (size_t)D * D_IN;               // [512][512]
  u16* WTq = WTk + (size_t)D * D;
  u16* WTv = WTq + (size_t)D * D;
  u16* e   = WTv + (size_t)D * D;                  // [M][512]
  u16* qb  = e  + (size_t)M * D;                   // [M][512]
  u16* kb  = qb + (size_t)M * D;                   // [M][512]
  u16* vT  = kb + (size_t)M * D;                   // [B][512][S]

  // 0) transpose + convert weights to bf16
  wt_transpose_kernel<<<(D_IN * D + 255) / 256, 256, 0, stream>>>(We, WTe, D_IN, D);
  wt_transpose_kernel<<<(D * D + 255) / 256, 256, 0, stream>>>(Wk, WTk, D, D);
  wt_transpose_kernel<<<(D * D + 255) / 256, 256, 0, stream>>>(Wq, WTq, D, D);
  wt_transpose_kernel<<<(D * D + 255) / 256, 256, 0, stream>>>(Wv, WTv, D, D);

  // 1) e = x @ W_embed   (f32 A converted on load)
  gemm_bf16_kernel<true, false><<<dim3(M / 64, D / 64), 256, 0, stream>>>(
      x, WTe, e, M, D, D_IN);

  // 2) q, k, vT = e @ W_{q,k,v}
  gemm_bf16_kernel<false, false><<<dim3(M / 64, D / 64), 256, 0, stream>>>(
      e, WTq, qb, M, D, D);
  gemm_bf16_kernel<false, false><<<dim3(M / 64, D / 64), 256, 0, stream>>>(
      e, WTk, kb, M, D, D);
  gemm_bf16_kernel<false, true><<<dim3(M / 64, D / 64), 256, 0, stream>>>(
      e, WTv, vT, M, D, D);

  // 3) flash attention (unscaled scores, per reference)
  attn_flash_kernel<<<dim3(S / 32, B), 256, 0, stream>>>(qb, kb, vT, out);
}
